// GRIL_7060926235270
// MI455X (gfx1250) — matrix operand order repaired and checked by reading, not compiled, not run on hardware
//
#include <hip/hip_runtime.h>
#include <hip/hip_bf16.h>
#include <cstddef>

// ---------------- problem constants ----------------
#define B_   2
#define NN   1024
#define T_   12
#define H_   64
#define CP   80     // channel dim padded to 5x16 for WMMA M-tiling
#define WT   128    // output-node tile per block in diffusion
#define KT   32     // K chunk (bf16 WMMA K)

typedef unsigned short u16;
typedef __attribute__((ext_vector_type(16))) __bf16 v16bf;
typedef __attribute__((ext_vector_type(8)))  float  v8f;
typedef __attribute__((address_space(3))) unsigned short lds_us;

union Frag16 { v16bf v; unsigned u[8]; };

static __device__ __forceinline__ u16 f2bf(float f) {
    union { float f; unsigned u; } c; c.f = f;
    unsigned r = c.u + 0x7FFFu + ((c.u >> 16) & 1u);   // RNE
    return (u16)(r >> 16);
}
static __device__ __forceinline__ float bf2f(u16 h) {
    union { unsigned u; float f; } c; c.u = ((unsigned)h) << 16;
    return c.f;
}

// =====================================================================
// Diffusion (all-bf16 operands): Y[bs][c][w] = sum_v X[.][c][v]*A[s][b][w][v]
//   X: (B,CP,N) bf16 (x_per_support=0) or (B,2,CP,N) bf16 (x_per_support=1)
//   A: (2,B,N,N) bf16 (pre-converted, L2 resident)
//   Y: (B,2,CP,N) bf16
// One block = 80x128 output tile; 8 wave32, wave w owns one 16-wide N slab.
// LDS double-buffered; tiles staged with GLOBAL_LOAD_ASYNC_TO_LDS_B128 and
// drained with s_wait_asynccnt so DMA of chunk k+1 overlaps WMMA of chunk k.
// =====================================================================
__global__ __launch_bounds__(256)
void gril_diffuse(const u16* __restrict__ X, const u16* __restrict__ A,
                  u16* __restrict__ Y, int x_per_support)
{
    __shared__ u16 ldsX[2][CP][KT];   // 2 x 80x32 bf16
    __shared__ u16 ldsA[2][WT][KT];   // 2 x 128x32 bf16

    const int tid  = threadIdx.x;
    const int wid  = tid >> 5;
    const int lane = tid & 31;
    const int wt   = blockIdx.x;          // 0..N/WT-1
    const int bs   = blockIdx.y;          // b*2+s
    const int b    = bs >> 1;
    const int s    = bs & 1;

    const u16* Xp = X + (size_t)(x_per_support ? bs : b) * CP * NN;
    const u16* Ap = A + ((size_t)s * B_ + b) * NN * NN + (size_t)wt * WT * NN;
    u16*       Yp = Y + (size_t)bs * CP * NN + (size_t)wt * WT;

    // issue async DMA for one K-chunk into LDS buffer `bufi`
    auto stage = [&](int bufi, int kv) {
        // X tile: 80 rows x 32 bf16 = 320 x 16B transfers
        for (int i = tid; i < (CP * KT) / 8; i += 256) {
            int row = i >> 2, seg = i & 3;
            const u16* gp = Xp + (size_t)row * NN + kv + seg * 8;
            unsigned lp = (unsigned)(size_t)(lds_us*)&ldsX[bufi][row][seg * 8];
            asm volatile("global_load_async_to_lds_b128 %0, %1, off"
                         :: "v"(lp), "v"(gp) : "memory");
        }
        // A tile: 128 rows x 32 bf16 = 512 x 16B transfers
        for (int i = tid; i < (WT * KT) / 8; i += 256) {
            int row = i >> 2, seg = i & 3;
            const u16* gp = Ap + (size_t)row * NN + kv + seg * 8;
            unsigned lp = (unsigned)(size_t)(lds_us*)&ldsA[bufi][row][seg * 8];
            asm volatile("global_load_async_to_lds_b128 %0, %1, off"
                         :: "v"(lp), "v"(gp) : "memory");
        }
    };

    v8f acc[5] = {};
    const int row16 = lane & 15;
    const int half  = lane >> 4;

    stage(0, 0);
    const int NCH = NN / KT;
    for (int kc = 0; kc < NCH; ++kc) {
        const int cur = kc & 1;
        asm volatile("s_wait_asynccnt 0" ::: "memory");  // my DMAs landed
        __syncthreads();                                 // everyone's landed
        if (kc + 1 < NCH) stage(cur ^ 1, (kc + 1) * KT); // prefetch next chunk

        // B fragment (32x16 = A^T slab): lanes 0-15 K=0..15, lanes 16-31 K=16..31
        Frag16 bf;
        {
            const int col = wid * 16 + row16;
#pragma unroll
            for (int i = 0; i < 8; ++i) {
                int k2 = half * 16 + i * 2;
                bf.u[i] = *(const unsigned*)&ldsA[cur][col][k2];
            }
        }
        // 5 M-tiles of X, one WMMA each
#pragma unroll
        for (int m = 0; m < 5; ++m) {
            Frag16 af;
            const int row = m * 16 + row16;
#pragma unroll
            for (int i = 0; i < 8; ++i) {
                int k2 = half * 16 + i * 2;
                af.u[i] = *(const unsigned*)&ldsX[cur][row][k2];
            }
            acc[m] = __builtin_amdgcn_wmma_f32_16x16x32_bf16(
                false, af.v, false, bf.v, (short)0, acc[m], false, false);
        }
    }

    // C/D layout: VGPR r -> M=r (lanes 0-15) / M=8+r (lanes 16-31), N=lane&15
    const int nloc = wid * 16 + row16;
    const int mofs = (lane < 16) ? 0 : 8;
#pragma unroll
    for (int m = 0; m < 5; ++m)
#pragma unroll
        for (int r = 0; r < 8; ++r)
            Yp[(size_t)(m * 16 + mofs + r) * NN + nloc] = f2bf(acc[m][r]);
}

// =====================================================================
// Adjacency f32 -> bf16 (once per launch; 4M elements, vectorized x4)
// =====================================================================
__global__ void gril_cvt_adj(const float* __restrict__ a, u16* __restrict__ abf)
{
    int idx = blockIdx.x * blockDim.x + threadIdx.x;   // element-quad index
    float4 v = ((const float4*)a)[idx];
    unsigned lo = (unsigned)f2bf(v.x) | ((unsigned)f2bf(v.y) << 16);
    unsigned hi = (unsigned)f2bf(v.z) | ((unsigned)f2bf(v.w) << 16);
    ((uint2*)abf)[idx] = make_uint2(lo, hi);
}

__global__ void gril_zero(float* p, int n) {
    int i = blockIdx.x * blockDim.x + threadIdx.x;
    if (i < n) p[i] = 0.f;
}

// =====================================================================
// Pointwise / per-node matvec kernels (one thread per (b,n))
// =====================================================================
__global__ void gril_pre(const float* __restrict__ x, const int* __restrict__ mask,
                         const float* __restrict__ h,
                         const float* __restrict__ Wfs, const float* __restrict__ bfs,
                         const float* __restrict__ Wdin, const float* __restrict__ bdin,
                         float* __restrict__ x1buf, u16* __restrict__ xin,
                         float* __restrict__ preds, int t)
{
    int idx = blockIdx.x * blockDim.x + threadIdx.x;
    int b = idx >> 10, n = idx & (NN - 1);
    float hv[H_];
    for (int c = 0; c < H_; ++c) hv[c] = h[((size_t)b * H_ + c) * NN + n];
    float xs1 = bfs[0];
    for (int c = 0; c < H_; ++c) xs1 += Wfs[c] * hv[c];
    size_t tix = ((size_t)b * NN + n) * T_ + t;
    preds[tix] = xs1;
    int m = mask[tix];
    float x1 = m ? x[tix] : xs1;
    x1buf[b * NN + n] = x1;
    float mf = (float)m;
    for (int c = 0; c < H_; ++c) {
        float a = bdin[c] + Wdin[c * 66] * x1 + Wdin[c * 66 + 1] * mf;
        for (int j = 0; j < H_; ++j) a += Wdin[c * 66 + 2 + j] * hv[j];
        xin[((size_t)b * CP + c) * NN + n] = f2bf(a);
    }
    for (int c = H_; c < CP; ++c) xin[((size_t)b * CP + c) * NN + n] = 0;
}

__global__ void gril_decpost(const int* __restrict__ mask, const float* __restrict__ h,
                             const u16* __restrict__ dec,
                             const float* __restrict__ Wgc, const float* __restrict__ bgc,
                             const float* __restrict__ Wlout, const float* __restrict__ blout,
                             const float* __restrict__ Wro, const float* __restrict__ bro,
                             const float* __restrict__ prelu_a,
                             const float* __restrict__ x1buf, float* __restrict__ x2buf,
                             u16* __restrict__ xh,
                             float* __restrict__ imps, float* __restrict__ reprs, int t)
{
    int idx = blockIdx.x * blockDim.x + threadIdx.x;
    int b = idx >> 10, n = idx & (NN - 1);
    float hv[H_], d[2 * H_], g[H_], o[H_];
    for (int c = 0; c < H_; ++c) hv[c] = h[((size_t)b * H_ + c) * NN + n];
    for (int c = 0; c < H_; ++c) {
        d[c]      = bf2f(dec[((size_t)(b * 2 + 0) * CP + c) * NN + n]);
        d[H_ + c] = bf2f(dec[((size_t)(b * 2 + 1) * CP + c) * NN + n]);
    }
    for (int c = 0; c < H_; ++c) {
        float a = bgc[c];
        for (int j = 0; j < 2 * H_; ++j) a += Wgc[c * 128 + j] * d[j];
        g[c] = a;
    }
    const float pa = prelu_a[0];
    for (int c = 0; c < H_; ++c) {
        float a = blout[c];
        for (int j = 0; j < H_; ++j) a += Wlout[c * 128 + j] * g[j];
        for (int j = 0; j < H_; ++j) a += Wlout[c * 128 + H_ + j] * hv[j];
        o[c] = (a >= 0.f) ? a : pa * a;
    }
    for (int c = 0; c < H_; ++c) {
        reprs[((size_t)(b * 128 + c) * NN + n) * T_ + t]      = o[c];
        reprs[((size_t)(b * 128 + H_ + c) * NN + n) * T_ + t] = hv[c];
    }
    float xs2 = bro[0];
    for (int j = 0; j < H_; ++j) xs2 += Wro[j] * o[j];
    for (int j = 0; j < H_; ++j) xs2 += Wro[H_ + j] * hv[j];
    size_t tix = ((size_t)b * NN + n) * T_ + t;
    imps[tix] = xs2;
    int m = mask[tix];
    float x1 = x1buf[b * NN + n];
    float x2 = m ? x1 : xs2;
    x2buf[b * NN + n] = x2;
    xh[((size_t)b * CP + 0) * NN + n] = f2bf(x2);
    xh[((size_t)b * CP + 1) * NN + n] = f2bf((float)m);
    for (int c = 0; c < H_; ++c) xh[((size_t)b * CP + 2 + c) * NN + n] = f2bf(hv[c]);
    for (int c = 2 + H_; c < CP; ++c) xh[((size_t)b * CP + c) * NN + n] = 0;
}

__global__ void gril_gates(const int* __restrict__ mask, const float* __restrict__ h,
                           const u16* __restrict__ xh, const u16* __restrict__ y1,
                           const u16* __restrict__ y2,
                           const float* __restrict__ Wr, const float* __restrict__ br,
                           const float* __restrict__ Wu, const float* __restrict__ bu,
                           const float* __restrict__ x2buf,
                           u16* __restrict__ xc, float* __restrict__ ubuf, int t)
{
    int idx = blockIdx.x * blockDim.x + threadIdx.x;
    int b = idx >> 10, n = idx & (NN - 1);
    float ra[H_], ua[H_];
    for (int c = 0; c < H_; ++c) { ra[c] = br[c]; ua[c] = bu[c]; }
    const u16* seg[5] = { xh + (size_t)b * CP * NN,
                          y1 + (size_t)(b * 2 + 0) * CP * NN,
                          y2 + (size_t)(b * 2 + 0) * CP * NN,
                          y1 + (size_t)(b * 2 + 1) * CP * NN,
                          y2 + (size_t)(b * 2 + 1) * CP * NN };
    for (int sgi = 0; sgi < 5; ++sgi) {
        const u16* bp = seg[sgi];
        for (int jj = 0; jj < 66; ++jj) {
            float zj = bf2f(bp[(size_t)jj * NN + n]);
            int j = sgi * 66 + jj;
            for (int c = 0; c < H_; ++c) {
                ra[c] += Wr[c * 330 + j] * zj;
                ua[c] += Wu[c * 330 + j] * zj;
            }
        }
    }
    for (int c = 0; c < H_; ++c) {
        ra[c] = 1.f / (1.f + __expf(-ra[c]));
        ua[c] = 1.f / (1.f + __expf(-ua[c]));
        ubuf[((size_t)b * H_ + c) * NN + n] = ua[c];
    }
    size_t tix = ((size_t)b * NN + n) * T_ + t;
    float mf = (float)mask[tix];
    xc[((size_t)b * CP + 0) * NN + n] = f2bf(x2buf[b * NN + n]);
    xc[((size_t)b * CP + 1) * NN + n] = f2bf(mf);
    for (int c = 0; c < H_; ++c)
        xc[((size_t)b * CP + 2 + c) * NN + n] =
            f2bf(ra[c] * h[((size_t)b * H_ + c) * NN + n]);
    for (int c = 2 + H_; c < CP; ++c) xc[((size_t)b * CP + c) * NN + n] = 0;
}

__global__ void gril_cell(const u16* __restrict__ xc, const u16* __restrict__ y1c,
                          const u16* __restrict__ y2c,
                          const float* __restrict__ Wc, const float* __restrict__ bc,
                          const float* __restrict__ ubuf, float* __restrict__ h)
{
    int idx = blockIdx.x * blockDim.x + threadIdx.x;
    int b = idx >> 10, n = idx & (NN - 1);
    float ca[H_];
    for (int c = 0; c < H_; ++c) ca[c] = bc[c];
    const u16* seg[5] = { xc  + (size_t)b * CP * NN,
                          y1c + (size_t)(b * 2 + 0) * CP * NN,
                          y2c + (size_t)(b * 2 + 0) * CP * NN,
                          y1c + (size_t)(b * 2 + 1) * CP * NN,
                          y2c + (size_t)(b * 2 + 1) * CP * NN };
    for (int sgi = 0; sgi < 5; ++sgi) {
        const u16* bp = seg[sgi];
        for (int jj = 0; jj < 66; ++jj) {
            float zj = bf2f(bp[(size_t)jj * NN + n]);
            int j = sgi * 66 + jj;
            for (int c = 0; c < H_; ++c) ca[c] += Wc[c * 330 + j] * zj;
        }
    }
    for (int c = 0; c < H_; ++c) {
        float cv = tanhf(ca[c]);
        size_t hx = ((size_t)b * H_ + c) * NN + n;
        float u = ubuf[hx];
        h[hx] = u * h[hx] + (1.f - u) * cv;
    }
}

// =====================================================================
// Host-side orchestration
// =====================================================================
extern "C" void kernel_launch(void* const* d_in, const int* in_sizes, int n_in,
                              void* d_out, int out_size, void* d_ws, size_t ws_size,
                              hipStream_t stream)
{
    const float* x      = (const float*)d_in[0];
    const int*   mask   = (const int*)  d_in[1];
    const float* adj    = (const float*)d_in[2];
    const float* Wr     = (const float*)d_in[3];
    const float* br     = (const float*)d_in[4];
    const float* Wu     = (const float*)d_in[5];
    const float* bu     = (const float*)d_in[6];
    const float* Wc     = (const float*)d_in[7];
    const float* bc     = (const float*)d_in[8];
    const float* Wfs    = (const float*)d_in[9];
    const float* bfs    = (const float*)d_in[10];
    const float* Wdin   = (const float*)d_in[11];
    const float* bdin   = (const float*)d_in[12];
    const float* Wgc    = (const float*)d_in[13];
    const float* bgc    = (const float*)d_in[14];
    const float* Wlout  = (const float*)d_in[15];
    const float* blout  = (const float*)d_in[16];
    const float* Wro    = (const float*)d_in[17];
    const float* bro    = (const float*)d_in[18];
    const float* prelua = (const float*)d_in[19];

    float* out   = (float*)d_out;
    float* imps  = out;                                   // (B,F,N,T)
    float* preds = out + (size_t)B_ * NN * T_;            // +24576
    float* reprs = out + (size_t)2 * B_ * NN * T_;        // +49152

    // ---- workspace layout (byte offsets, all 16B aligned) ----
    char* wsb = (char*)d_ws;
    size_t off = 0;
    float* h     = (float*)(wsb + off); off += (size_t)B_ * H_ * NN * 4;       // 512K
    float* x1buf = (float*)(wsb + off); off += (size_t)B_ * NN * 4;
    float* x2buf = (float*)(wsb + off); off += (size_t)B_ * NN * 4;
    float* ubuf  = (float*)(wsb + off); off += (size_t)B_ * H_ * NN * 4;
    u16*  adjbf  = (u16*)(wsb + off);   off += (size_t)2 * B_ * NN * NN * 2;   // 8M
    u16*  xin    = (u16*)(wsb + off);   off += (size_t)B_ * CP * NN * 2;
    u16*  xh     = (u16*)(wsb + off);   off += (size_t)B_ * CP * NN * 2;
    u16*  xc     = (u16*)(wsb + off);   off += (size_t)B_ * CP * NN * 2;
    u16*  dec    = (u16*)(wsb + off);   off += (size_t)B_ * 2 * CP * NN * 2;
    u16*  y1     = (u16*)(wsb + off);   off += (size_t)B_ * 2 * CP * NN * 2;
    u16*  y2     = (u16*)(wsb + off);   off += (size_t)B_ * 2 * CP * NN * 2;
    u16*  y1c    = (u16*)(wsb + off);   off += (size_t)B_ * 2 * CP * NN * 2;
    u16*  y2c    = (u16*)(wsb + off);   off += (size_t)B_ * 2 * CP * NN * 2;

    // adjacency -> bf16 (4M elements, 4 per thread)
    {
        int nq = (2 * B_ * NN * NN) / 4;
        gril_cvt_adj<<<nq / 256, 256, 0, stream>>>(adj, adjbf);
    }
    // h0 = 0
    {
        int nh = B_ * H_ * NN;
        gril_zero<<<(nh + 255) / 256, 256, 0, stream>>>(h, nh);
    }

    const dim3 dgrid(NN / WT, B_ * 2);   // (8,4)
    const int  PW_BLK = 128;
    const int  PW_GRD = (B_ * NN) / PW_BLK;

    for (int t = 0; t < T_; ++t) {
        gril_pre<<<PW_GRD, PW_BLK, 0, stream>>>(x, mask, h, Wfs, bfs, Wdin, bdin,
                                                x1buf, xin, preds, t);
        gril_diffuse<<<dgrid, 256, 0, stream>>>(xin, adjbf, dec, 0);
        gril_decpost<<<PW_GRD, PW_BLK, 0, stream>>>(mask, h, dec, Wgc, bgc, Wlout, blout,
                                                    Wro, bro, prelua, x1buf, x2buf, xh,
                                                    imps, reprs, t);
        gril_diffuse<<<dgrid, 256, 0, stream>>>(xh, adjbf, y1, 0);
        gril_diffuse<<<dgrid, 256, 0, stream>>>(y1, adjbf, y2, 1);
        gril_gates<<<PW_GRD, PW_BLK, 0, stream>>>(mask, h, xh, y1, y2, Wr, br, Wu, bu,
                                                  x2buf, xc, ubuf, t);
        gril_diffuse<<<dgrid, 256, 0, stream>>>(xc, adjbf, y1c, 0);
        gril_diffuse<<<dgrid, 256, 0, stream>>>(y1c, adjbf, y2c, 1);
        gril_cell<<<PW_GRD, PW_BLK, 0, stream>>>(xc, y1c, y2c, Wc, bc, ubuf, h);
    }
}
